// SawSub_27676769255637
// MI455X (gfx1250) — matrix operand order repaired and checked by reading, not compiled, not run on hardware
//
#include <hip/hip_runtime.h>
#include <hip/hip_bf16.h>
#include <math.h>
#include <stdint.h>

// ---------------------------------------------------------------------------
// DDSP-style synth pipeline for gfx1250 (MI455X), wave32 + WMMA.
//   MLP (4 WMMA GEMMs) -> controls -> sawtooth (closed-form scan)
//   -> IR synthesis AS A WMMA GEMM (irfft/fftshift/hann = fixed linear basis)
//   -> time-domain conv + overlap-add
// ---------------------------------------------------------------------------

#define SR 24000.0f
#define TWO_PI 6.283185307179586f
#define NB 8          // batch
#define NF 1000       // frames per batch
#define FTOT (NB*NF)  // 8000
#define HOP 240
#define T_SAMP (NF*HOP)       // 240000
#define KIN 80
#define KINP 96
#define H 256
#define NOUT 593
#define NOUTP 608
#define NMAG_H 512
#define IR_H 1022     // 2*(512-1)
#define IRP_H 1024    // padded N for WMMA
#define START_H 509   // (1022-1)//2 - 1
#define NMAG_N 80
#define NMAG_NP 96
#define IR_N 158      // 2*(80-1)
#define IRP_N 160
#define START_N 77    // (158-1)//2 - 1

typedef _Float16 h8   __attribute__((ext_vector_type(8)));
typedef _Float16 v16h __attribute__((ext_vector_type(16)));
typedef float    v8f  __attribute__((ext_vector_type(8)));

__device__ __forceinline__ float sigm(float x) { return 1.0f / (1.0f + __expf(-x)); }
__device__ __forceinline__ float geluf(float x) {
  const float c = 0.7978845608028654f;
  return 0.5f * x * (1.0f + tanhf(c * (x + 0.044715f * x * x * x)));
}
__device__ __forceinline__ float exp_sigmoid(float x) {
  return 2.0f * __powf(sigm(x), 2.302585092994046f) + 1e-7f;
}

// ------------------------- conversion kernels ------------------------------

__global__ void cvt_mel_kernel(const float* __restrict__ mel, _Float16* __restrict__ out) {
  int idx = blockIdx.x * blockDim.x + threadIdx.x;
  if (idx >= FTOT * KINP) return;
  int r = idx / KINP, c = idx % KINP;
  float v = (c < KIN) ? mel[(size_t)r * KIN + c] : 0.0f;
  out[idx] = (_Float16)v;
}

// W [K][N] f32 -> Wt [Npad][Kpad] f16 (transpose + zero pad)
__global__ void cvt_wt_kernel(const float* __restrict__ W, _Float16* __restrict__ Wt,
                              int K, int N, int Kpad, int Npad) {
  int idx = blockIdx.x * blockDim.x + threadIdx.x;
  if (idx >= Npad * Kpad) return;
  int n = idx / Kpad, k = idx % Kpad;
  float v = (k < K && n < N) ? W[(size_t)k * N + n] : 0.0f;
  Wt[idx] = (_Float16)v;
}

__global__ void pad_bias_kernel(const float* __restrict__ b, float* __restrict__ bp) {
  int idx = blockIdx.x * blockDim.x + threadIdx.x;
  if (idx >= NOUTP) return;
  bp[idx] = (idx < NOUT) ? b[idx] : 0.0f;
}

__global__ void zero_kernel(float* __restrict__ p, int n) {
  int idx = blockIdx.x * blockDim.x + threadIdx.x;
  if (idx < n) p[idx] = 0.0f;
}

// IR synthesis basis: ir[f][n] = sum_k mag[f][k] * B[k][n]
//   B[k][n] = hann(n)/N * c_k * cos(2*pi*k*shift(n)/N), shift(n)=(n+N/2)%N,
//   c_0 = 1, c_{nm-1} = +-1 (Nyquist), else 2.  Stored transposed [Npad][Kpad] f16.
__global__ void basis_kernel(_Float16* __restrict__ Bt, int nmags, int irs,
                             int Kpad, int Npad) {
  int idx = blockIdx.x * blockDim.x + threadIdx.x;
  if (idx >= Npad * Kpad) return;
  int n = idx / Kpad, k = idx % Kpad;
  float v = 0.0f;
  if (n < irs && k < nmags) {
    int half = irs >> 1;
    int src  = n + half; if (src >= irs) src -= irs;
    float w  = TWO_PI / (float)irs;
    float hn = 0.5f * (1.0f - cosf(w * (float)n)) / (float)irs;
    if (k == 0)                 v = hn;
    else if (k == nmags - 1)    v = hn * ((src & 1) ? -1.0f : 1.0f);
    else {
      int km = (k * src) % irs;             // keep cos arg in [0, 2pi)
      v = 2.0f * hn * cosf(w * (float)km);
    }
  }
  Bt[idx] = (_Float16)v;
}

// ----------------------------- WMMA GEMM -----------------------------------
// One 16xK A-fragment per wave, reused across NT adjacent 16-wide N-tiles.
// All NT B-fragments are staged into registers before the WMMA chain so the
// compiler can issue one load clause and use partial load-count waits.
// MODE 0: out = tanh(A@W+b)     -> f16
// MODE 1: out = A + gelu(A@W+b) -> f16   (residual == A)
// MODE 2: out = A@W+b           -> f32
template <int MODE, int NT>
__global__ void wmma_gemm_kernel(const _Float16* __restrict__ A, int lda,
                                 const _Float16* __restrict__ Wt, int ldw,
                                 const float* __restrict__ bias,
                                 _Float16* __restrict__ OutH,
                                 float* __restrict__ OutF,
                                 int ldo, int K) {
  const int wave   = threadIdx.x >> 5;
  const int lane   = threadIdx.x & 31;
  const int tileM  = blockIdx.x * (blockDim.x >> 5) + wave;
  const int tileN0 = blockIdx.y * NT;
  const int r15    = lane & 15;
  const int hi     = lane >> 4;                // half-wave select
  const int akb    = hi * 8;                   // A k-base (ISA 16-bit A layout)
  const int bkb    = hi * 8;                   // B k-base (same k order as A)

  const _Float16* __restrict__ Arow = A + (size_t)(tileM * 16 + r15) * lda;
  const _Float16* __restrict__ Brow[NT];
#pragma unroll
  for (int t = 0; t < NT; ++t)
    Brow[t] = Wt + (size_t)((tileN0 + t) * 16 + r15) * ldw;

  v8f c[NT];
#pragma unroll
  for (int t = 0; t < NT; ++t) c[t] = v8f{};

  for (int k0 = 0; k0 < K; k0 += 32) {
    // --- issue ALL loads for this k-step first ---
    h8 a_lo = *(const h8*)(Arow + k0 + akb);
    h8 a_hi = *(const h8*)(Arow + k0 + 16 + akb);
    h8 bl0[NT], bl1[NT];
#pragma unroll
    for (int t = 0; t < NT; ++t) {
      bl0[t] = *(const h8*)(Brow[t] + k0 + bkb);
      bl1[t] = *(const h8*)(Brow[t] + k0 + 16 + bkb);
    }
    v16h a;
#pragma unroll
    for (int i = 0; i < 8; ++i) { a[i] = a_lo[i]; a[8 + i] = a_hi[i]; }
    // --- WMMA chain, one per N-tile, all sharing the A fragment ---
#pragma unroll
    for (int t = 0; t < NT; ++t) {
      v16h b;
#pragma unroll
      for (int i = 0; i < 8; ++i) { b[i] = bl0[t][i]; b[8 + i] = bl1[t][i]; }
      c[t] = __builtin_amdgcn_wmma_f32_16x16x32_f16(false, a, false, b,
                                                    (short)0, c[t], false, false);
    }
  }

  const int orow0 = tileM * 16 + hi * 8;       // C/D layout: VGPR v -> row
#pragma unroll
  for (int t = 0; t < NT; ++t) {
    const int ocol = (tileN0 + t) * 16 + r15;
    const float bv = bias[ocol];
#pragma unroll
    for (int v = 0; v < 8; ++v) {
      const size_t oidx = (size_t)(orow0 + v) * ldo + ocol;
      float z = c[t][v] + bv;
      if (MODE == 0) {
        OutH[oidx] = (_Float16)tanhf(z);
      } else if (MODE == 1) {
        float hprev = (float)A[(size_t)(orow0 + v) * lda + ocol];
        OutH[oidx] = (_Float16)(hprev + geluf(z));
      } else {
        OutF[oidx] = z;
      }
    }
  }
}

// ----------------------------- controls ------------------------------------
// Writes f0 (f32) and the filter magnitudes directly in f16 (padded) so they
// feed the IR-synthesis WMMA GEMM without a conversion pass.
__global__ void ctrl_kernel(const float* __restrict__ o,
                            float* __restrict__ f0_dout, float* __restrict__ f0_ws,
                            _Float16* __restrict__ srcPh, _Float16* __restrict__ noiPh) {
  int idx = blockIdx.x * blockDim.x + threadIdx.x;
  if (idx >= FTOT * NOUT) return;
  int f = idx / NOUT, c = idx % NOUT;
  float x = o[(size_t)f * NOUTP + c];
  if (c == 0) {
    float u  = sigm(x);
    float f0 = 80.0f * __expf(u * 2.5257286443082556f);  // 80*(12.5)^u
    if (f0 < 80.0f) f0 = 0.0f;
    f0_dout[f] = f0;
    f0_ws[f]   = f0;
  } else if (c < 513) {
    srcPh[(size_t)f * NMAG_H + (c - 1)] = (_Float16)exp_sigmoid(x);
  } else {
    noiPh[(size_t)f * NMAG_NP + (c - 513)] = (_Float16)exp_sigmoid(x);
  }
}

__global__ void pad_noise_mags_kernel(_Float16* __restrict__ noiPh) {
  int idx = blockIdx.x * blockDim.x + threadIdx.x;
  if (idx >= FTOT * (NMAG_NP - NMAG_N)) return;
  int f = idx / (NMAG_NP - NMAG_N), j = idx % (NMAG_NP - NMAG_N);
  noiPh[(size_t)f * NMAG_NP + NMAG_N + j] = (_Float16)0.0f;
}

// exclusive prefix over per-frame phase sums (tiny: 8 lanes x 1000 iters)
__global__ void prefix_kernel(const float* __restrict__ f0, float* __restrict__ prefix) {
  int b = blockIdx.x * blockDim.x + threadIdx.x;
  if (b >= NB) return;
  const float* f = f0 + (size_t)b * NF;
  float acc = 0.0f;
  for (int i = 0; i < NF; ++i) {
    prefix[(size_t)b * NF + i] = acc;
    float fa = f[i];
    float fb = (i + 1 < NF) ? f[i + 1] : f[NF - 1];
    acc += 120.5f * fa + 119.5f * fb;   // closed-form frame sum of linear upsample
  }
}

__global__ void saw_kernel(const float* __restrict__ f0, const float* __restrict__ prefix,
                           const float* __restrict__ init_phase,
                           float* __restrict__ audio, float* __restrict__ final_phase) {
  int idx = blockIdx.x * blockDim.x + threadIdx.x;
  if (idx >= NB * T_SAMP) return;
  int b = idx / T_SAMP, s = idx % T_SAMP;
  int i = s / HOP, t = s % HOP;
  const float* f = f0 + (size_t)b * NF;
  float fa = f[i];
  float fb = (i + 1 < NF) ? f[i + 1] : f[NF - 1];
  // inclusive within-frame cumsum of the linear ramp, closed form
  float within = (float)(t + 1) * fa + (fb - fa) * ((float)(t * (t + 1)) * (0.5f / (float)HOP));
  float cum    = prefix[(size_t)b * NF + i] + within;
  float phase  = TWO_PI * cum * (1.0f / SR) + init_phase[b];
  float fr     = phase * (1.0f / TWO_PI);
  fr           = fr - floorf(fr);
  audio[idx]   = 2.0f * fr - 1.0f;
  if (s == T_SAMP - 1) final_phase[b] = fmodf(phase, TWO_PI);
}

// deterministic hash-based uniform noise in [-1, 1)
__device__ __forceinline__ float hash_u01(uint32_t x) {
  x ^= x >> 17; x *= 0xed5ad4bbu;
  x ^= x >> 11; x *= 0xac4c1b51u;
  x ^= x >> 15; x *= 0x31848babu;
  x ^= x >> 14;
  return (float)(x >> 8) * (1.0f / 16777216.0f);
}
__global__ void noise_kernel(float* __restrict__ out) {
  int idx = blockIdx.x * blockDim.x + threadIdx.x;
  if (idx >= NB * T_SAMP) return;
  out[idx] = 2.0f * hash_u01((uint32_t)idx * 2654435761u + 0x9E3779B9u) - 1.0f;
}

// --------------- time-varying FIR: conv + overlap-add (gather) -------------
__global__ void conv_ola_kernel(const float* __restrict__ audio, const float* __restrict__ ir,
                                int irs, int ld_ir, int start, float* __restrict__ out) {
  int idx = blockIdx.x * blockDim.x + threadIdx.x;
  if (idx >= NB * T_SAMP) return;
  int b = idx / T_SAMP, s = idx % T_SAMP;
  int sa  = s + start;
  int fhi = sa / HOP; if (fhi > NF - 1) fhi = NF - 1;
  int num = sa - (HOP + irs - 2);
  int flo = (num <= 0) ? 0 : (num + HOP - 1) / HOP;
  const float* au = audio + (size_t)b * T_SAMP;
  float acc = 0.0f;
  for (int f = flo; f <= fhi; ++f) {
    int q   = sa - f * HOP;
    int jlo = q - irs + 1; if (jlo < 0) jlo = 0;
    int jhi = q;           if (jhi > HOP - 1) jhi = HOP - 1;
    const float* irf = ir + ((size_t)b * NF + f) * ld_ir;
    if (f < fhi) __builtin_prefetch(irf + ld_ir, 0, 1);   // global_prefetch_b8
    const float* auf = au + (size_t)f * HOP;
    for (int j = jlo; j <= jhi; ++j)
      acc += auf[j] * irf[q - j];
  }
  out[idx] = acc;
}

__global__ void combine_kernel(const float* __restrict__ ha, const float* __restrict__ no,
                               float* __restrict__ sig) {
  int idx = blockIdx.x * blockDim.x + threadIdx.x;
  if (idx >= NB * T_SAMP) return;
  sig[idx] = ha[idx] + no[idx];
}

// ------------------------------- launch ------------------------------------

static inline int cdiv(int a, int b) { return (a + b - 1) / b; }

extern "C" void kernel_launch(void* const* d_in, const int* in_sizes, int n_in,
                              void* d_out, int out_size, void* d_ws, size_t ws_size,
                              hipStream_t stream) {
  const float* mel    = (const float*)d_in[0];
  const float* iphase = (const float*)d_in[1];
  const float* W_in   = (const float*)d_in[2];
  const float* b_in   = (const float*)d_in[3];
  const float* W1     = (const float*)d_in[4];
  const float* b1     = (const float*)d_in[5];
  const float* W2     = (const float*)d_in[6];
  const float* b2     = (const float*)d_in[7];
  const float* W_out  = (const float*)d_in[8];
  const float* b_out  = (const float*)d_in[9];

  float* out_signal = (float*)d_out;
  float* out_f0     = out_signal + (size_t)NB * T_SAMP;
  float* out_fp     = out_f0 + FTOT;
  float* out_harm   = out_fp + NB;
  float* out_noise  = out_harm + (size_t)NB * T_SAMP;

  // workspace bump allocator (256B aligned)
  char* p = (char*)d_ws;
  auto alloc = [&](size_t bytes) -> void* {
    void* r = (void*)p;
    p += (bytes + 255) & ~(size_t)255;
    return r;
  };
  _Float16* melp   = (_Float16*)alloc((size_t)FTOT * KINP * 2);
  _Float16* WtIn   = (_Float16*)alloc((size_t)H * KINP * 2);
  _Float16* Wt1    = (_Float16*)alloc((size_t)H * H * 2);
  _Float16* Wt2    = (_Float16*)alloc((size_t)H * H * 2);
  _Float16* WtOut  = (_Float16*)alloc((size_t)NOUTP * H * 2);
  float*    bOutP  = (float*)alloc((size_t)NOUTP * 4);
  float*    zbias  = (float*)alloc((size_t)IRP_H * 4);
  _Float16* h0     = (_Float16*)alloc((size_t)FTOT * H * 2);
  _Float16* h1     = (_Float16*)alloc((size_t)FTOT * H * 2);
  _Float16* h2     = (_Float16*)alloc((size_t)FTOT * H * 2);
  float*    obuf   = (float*)alloc((size_t)FTOT * NOUTP * 4);
  float*    f0v    = (float*)alloc((size_t)FTOT * 4);
  _Float16* srcPh  = (_Float16*)alloc((size_t)FTOT * NMAG_H * 2);
  _Float16* noiPh  = (_Float16*)alloc((size_t)FTOT * NMAG_NP * 2);
  _Float16* basisH = (_Float16*)alloc((size_t)IRP_H * NMAG_H * 2);
  _Float16* basisN = (_Float16*)alloc((size_t)IRP_N * NMAG_NP * 2);
  float*    prefix = (float*)alloc((size_t)FTOT * 4);
  float*    rawH   = (float*)alloc((size_t)NB * T_SAMP * 4);
  float*    rawN   = (float*)alloc((size_t)NB * T_SAMP * 4);
  float*    irH    = (float*)alloc((size_t)FTOT * IRP_H * 4);
  float*    irN    = (float*)alloc((size_t)FTOT * IRP_N * 4);

  const int BT = 256;

  // ---- precision/layout conversion + constant tables ----
  cvt_mel_kernel<<<cdiv(FTOT * KINP, BT), BT, 0, stream>>>(mel, melp);
  cvt_wt_kernel<<<cdiv(H * KINP, BT), BT, 0, stream>>>(W_in, WtIn, KIN, H, KINP, H);
  cvt_wt_kernel<<<cdiv(H * H, BT), BT, 0, stream>>>(W1, Wt1, H, H, H, H);
  cvt_wt_kernel<<<cdiv(H * H, BT), BT, 0, stream>>>(W2, Wt2, H, H, H, H);
  cvt_wt_kernel<<<cdiv(NOUTP * H, BT), BT, 0, stream>>>(W_out, WtOut, H, NOUT, H, NOUTP);
  pad_bias_kernel<<<cdiv(NOUTP, BT), BT, 0, stream>>>(b_out, bOutP);
  zero_kernel<<<cdiv(IRP_H, BT), BT, 0, stream>>>(zbias, IRP_H);
  basis_kernel<<<cdiv(IRP_H * NMAG_H, BT), BT, 0, stream>>>(basisH, NMAG_H, IR_H, NMAG_H, IRP_H);
  basis_kernel<<<cdiv(IRP_N * NMAG_NP, BT), BT, 0, stream>>>(basisN, NMAG_N, IR_N, NMAG_NP, IRP_N);
  pad_noise_mags_kernel<<<cdiv(FTOT * (NMAG_NP - NMAG_N), BT), BT, 0, stream>>>(noiPh);

  // ---- MLP: 4 fused WMMA GEMMs (wave32, 4 waves/block, NT N-tiles/wave) ----
  dim3 blk(128);
  const int MB = FTOT / 16 / 4;         // 125 M-tile blocks
  wmma_gemm_kernel<0, 4><<<dim3(MB, (H / 16) / 4), blk, 0, stream>>>(
      melp, KINP, WtIn, KINP, b_in, h0, nullptr, H, KINP);
  wmma_gemm_kernel<1, 4><<<dim3(MB, (H / 16) / 4), blk, 0, stream>>>(
      h0, H, Wt1, H, b1, h1, nullptr, H, H);
  wmma_gemm_kernel<1, 4><<<dim3(MB, (H / 16) / 4), blk, 0, stream>>>(
      h1, H, Wt2, H, b2, h2, nullptr, H, H);
  wmma_gemm_kernel<2, 2><<<dim3(MB, (NOUTP / 16) / 2), blk, 0, stream>>>(
      h2, H, WtOut, H, bOutP, nullptr, obuf, NOUTP, H);

  // ---- controls ----
  ctrl_kernel<<<cdiv(FTOT * NOUT, BT), BT, 0, stream>>>(obuf, out_f0, f0v, srcPh, noiPh);

  // ---- sawtooth via closed-form scan ----
  prefix_kernel<<<1, 32, 0, stream>>>(f0v, prefix);
  saw_kernel<<<cdiv(NB * T_SAMP, BT), BT, 0, stream>>>(f0v, prefix, iphase, rawH, out_fp);
  noise_kernel<<<cdiv(NB * T_SAMP, BT), BT, 0, stream>>>(rawN);

  // ---- IR synthesis as WMMA GEMM: [8000 x K] @ [K x IRP] ----
  wmma_gemm_kernel<2, 4><<<dim3(MB, (IRP_H / 16) / 4), blk, 0, stream>>>(
      srcPh, NMAG_H, basisH, NMAG_H, zbias, nullptr, irH, IRP_H, NMAG_H);
  wmma_gemm_kernel<2, 2><<<dim3(MB, (IRP_N / 16) / 2), blk, 0, stream>>>(
      noiPh, NMAG_NP, basisN, NMAG_NP, zbias, nullptr, irN, IRP_N, NMAG_NP);

  // ---- time-varying FIR: conv + overlap-add ----
  conv_ola_kernel<<<cdiv(NB * T_SAMP, BT), BT, 0, stream>>>(rawH, irH, IR_H, IRP_H, START_H, out_harm);
  conv_ola_kernel<<<cdiv(NB * T_SAMP, BT), BT, 0, stream>>>(rawN, irN, IR_N, IRP_N, START_N, out_noise);

  combine_kernel<<<cdiv(NB * T_SAMP, BT), BT, 0, stream>>>(out_harm, out_noise, out_signal);

  (void)in_sizes; (void)n_in; (void)out_size; (void)ws_size;
}
